// MultiHeadAttention_31722628448375
// MI455X (gfx1250) — matrix operand order repaired and checked by reading, not compiled, not run on hardware
//
#include <hip/hip_runtime.h>
#include <hip/hip_bf16.h>

typedef __attribute__((ext_vector_type(16))) _Float16 v16h;
typedef __attribute__((ext_vector_type(8)))  _Float16 v8h;
typedef __attribute__((ext_vector_type(8)))  float    v8f;

#define B_ 4
#define N_ 2048
#define C_ 1024
#define H_ 16
#define D_ 64
#define SCALE_ 0.125f

// ---------------------------------------------------------------------------
// Packed fragment layout: frag f = base + (f*32 + lane)*16 halves (32B/lane,
// 1KB contiguous per wave-fragment -> two coalesced b128 loads per lane).
// A-fragment element e of lane (lr,g):  K = e + 8*g + (e>=8 ? 8 : 0)
// B-fragment element e of lane (lr,g):  K = e + 8*g + (e>=8 ? 8 : 0)
// ---------------------------------------------------------------------------
__device__ __forceinline__ v16h load_frag16(const _Float16* __restrict__ base,
                                            size_t frag, int lane) {
    const v8h* p = (const v8h*)(base + (frag * 32 + (size_t)lane) * 16);
    v8h lo = p[0], hi = p[1];
    return __builtin_shufflevector(lo, hi, 0, 1, 2, 3, 4, 5, 6, 7,
                                   8, 9, 10, 11, 12, 13, 14, 15);
}

#define WMMA_F16(A, Bv, Cv) \
    __builtin_amdgcn_wmma_f32_16x16x32_f16(false, (A), false, (Bv), (short)0, (Cv), false, false)

// K=1024 GEMM inner loop: 32 k-chunks, 4 B-tiles per wave, fully unrolled.
// sched_group_barrier imposes a software pipeline on the single scheduling
// region: 20 loads up front (2 chunks), then (4 WMMA, 10 loads) x 30, then
// the final 8 WMMA -- each chunk's loads are issued 2 wmma-groups early, so
// waits land at ~10 outstanding loads instead of draining to zero.
__device__ __forceinline__ void gemm_k32(const _Float16* __restrict__ Apk,
                                         const _Float16* __restrict__ Bpk,
                                         size_t af0, const size_t* bf0,
                                         int lane, v8f* acc) {
    v16h a[2], b[2][4];
    a[0] = load_frag16(Apk, af0 + 0, lane);
#pragma unroll
    for (int s = 0; s < 4; ++s) b[0][s] = load_frag16(Bpk, bf0[s] + 0, lane);
    a[1] = load_frag16(Apk, af0 + 1, lane);
#pragma unroll
    for (int s = 0; s < 4; ++s) b[1][s] = load_frag16(Bpk, bf0[s] + 1, lane);

#pragma unroll
    for (int kc = 0; kc < 32; ++kc) {
        const int cur = kc & 1;
#pragma unroll
        for (int s = 0; s < 4; ++s) acc[s] = WMMA_F16(a[cur], b[cur][s], acc[s]);
        if (kc + 2 < 32) {
            a[cur] = load_frag16(Apk, af0 + kc + 2, lane);
#pragma unroll
            for (int s = 0; s < 4; ++s) b[cur][s] = load_frag16(Bpk, bf0[s] + kc + 2, lane);
        }
    }

    // Pipeline shape: 0x020 = VMEM read group, 0x008 = MFMA/WMMA group.
    __builtin_amdgcn_sched_group_barrier(0x020, 20, 0);
#pragma unroll
    for (int kc = 0; kc < 30; ++kc) {
        __builtin_amdgcn_sched_group_barrier(0x008, 4, 0);
        __builtin_amdgcn_sched_group_barrier(0x020, 10, 0);
    }
    __builtin_amdgcn_sched_group_barrier(0x008, 8, 0);
}

// f32 [R x K] row-major -> packed A fragments (rtile-major, then k-chunk)
__global__ __launch_bounds__(256) void pack_a_f32(
    const float* __restrict__ in, _Float16* __restrict__ outp, int kchunks)
{
    const int frag = blockIdx.x * 8 + (threadIdx.x >> 5);
    const int lane = threadIdx.x & 31, lr = lane & 15, g = lane >> 4;
    const int rt = frag / kchunks, kc = frag % kchunks;
    const int K = kchunks * 32;
    const float* src = in + (size_t)(rt * 16 + lr) * K + kc * 32 + 8 * g;
    v8h lo, hi;
#pragma unroll
    for (int j = 0; j < 8; ++j) { lo[j] = (_Float16)src[j]; hi[j] = (_Float16)src[16 + j]; }
    v8h* dst = (v8h*)(outp + ((size_t)frag * 32 + lane) * 16);
    dst[0] = lo; dst[1] = hi;
}

// f32 [R x K] row-major -> packed B fragments (R is the GEMM N dim)
__global__ __launch_bounds__(256) void pack_b_f32(
    const float* __restrict__ in, _Float16* __restrict__ outp, int kchunks)
{
    const int frag = blockIdx.x * 8 + (threadIdx.x >> 5);
    const int lane = threadIdx.x & 31, lr = lane & 15, g = lane >> 4;
    const int rt = frag / kchunks, kc = frag % kchunks;
    const int K = kchunks * 32;
    const float* src = in + (size_t)(rt * 16 + lr) * K + kc * 32 + 8 * g;
    v8h lo, hi;
#pragma unroll
    for (int j = 0; j < 8; ++j) { lo[j] = (_Float16)src[j]; hi[j] = (_Float16)src[16 + j]; }
    v8h* dst = (v8h*)(outp + ((size_t)frag * 32 + lane) * 16);
    dst[0] = lo; dst[1] = hi;
}

// ---------------------------------------------------------------------------
// Kernel 1: fused QKV GEMM from packed x / packed [Wq;Wkv].
// Epilogue scatters into the attention kernel's packed fragment layouts:
//   Qpk: A-frags keyed ((b*H+h)*128 + ntile)*2 + d-chunk   (Q pre-scaled)
//   Kpk: B-frags keyed ((b*H+h)*128 + mtile)*2 + d-chunk
//   Vpk: B-frags keyed ((b*H+h)*4 + dtile)*64 + m-chunk    (V transposed)
// ---------------------------------------------------------------------------
__global__ __launch_bounds__(256) void qkv_kernel(
    const _Float16* __restrict__ xpk, const _Float16* __restrict__ wpk,
    const float* __restrict__ bq, const float* __restrict__ bkv,
    _Float16* __restrict__ Qpk, _Float16* __restrict__ Kpk, _Float16* __restrict__ Vpk)
{
    const int wid  = blockIdx.x * 8 + (threadIdx.x >> 5);
    const int lane = threadIdx.x & 31;
    const int lr   = lane & 15;
    const int g    = lane >> 4;
    const int cg   = wid % 48;   // 48 groups of 64 output cols (3072 total)
    const int mt   = wid / 48;   // 512 row tiles

    v8f acc[4];
#pragma unroll
    for (int s = 0; s < 4; ++s) acc[s] = (v8f){};

    const size_t af0 = (size_t)mt * 32;
    size_t bf0[4];
#pragma unroll
    for (int s = 0; s < 4; ++s) bf0[s] = (size_t)(cg * 4 + s) * 32;

    gemm_k32(xpk, wpk, af0, bf0, lane, acc);

    // epilogue: bias (+SCALE into Q), scatter to packed consumer layouts
#pragma unroll
    for (int s = 0; s < 4; ++s) {
        const int c = cg * 64 + s * 16 + lr;             // uniform branch per (wave,s)
        const float bias = (c < 1024) ? bq[c] : bkv[c - 1024];
        const float scl  = (c < 1024) ? SCALE_ : 1.0f;
#pragma unroll
        for (int r = 0; r < 8; ++r) {
            const int m = mt * 16 + r + 8 * g;
            const int b = m >> 11, n = m & (N_ - 1);
            const float v = (acc[s][r] + bias) * scl;
            if (c < 1024) {                               // Q -> packed A frags
                const int h = c >> 6, d = c & 63;
                const int ko = d & 31, gq = (ko >> 3) & 1;
                const int e = (ko & 7) + 8 * (ko >> 4);
                const size_t idx =
                    ((((size_t)(b * H_ + h) * 128 + (n >> 4)) * 2 + (d >> 5)) * 32 +
                     ((n & 15) + 16 * gq)) * 16 + e;
                Qpk[idx] = (_Float16)v;
            } else if (c < 2048) {                        // K -> packed B frags
                const int cc = c - 1024, h = cc >> 6, d = cc & 63;
                const size_t idx =
                    ((((size_t)(b * H_ + h) * 128 + (n >> 4)) * 2 + (d >> 5)) * 32 +
                     ((n & 15) + 16 * ((d >> 3) & 1))) * 16 + ((d & 7) + 8 * ((d >> 4) & 1));
                Kpk[idx] = (_Float16)v;
            } else {                                      // V -> packed B frags (k = m)
                const int cc = c - 2048, h = cc >> 6, d = cc & 63;
                const size_t idx =
                    ((((size_t)(b * H_ + h) * 4 + (d >> 4)) * 64 + (n >> 5)) * 32 +
                     ((d & 15) + 16 * ((n >> 3) & 1))) * 16 + ((n & 7) + 8 * ((n >> 4) & 1));
                Vpk[idx] = (_Float16)v;
            }
        }
    }
}

// ---------------------------------------------------------------------------
// Kernel 2: fused attention per (b, h, 16-row tile).
// ---------------------------------------------------------------------------
__global__ __launch_bounds__(256) void attn_kernel(
    const _Float16* __restrict__ Qpk, const _Float16* __restrict__ Kpk,
    const _Float16* __restrict__ Vpk,
    float* __restrict__ attn_out, _Float16* __restrict__ Opk)
{
    __shared__ float S[16 * 2048];     // 128 KB score/prob strip
    __shared__ float Op[8 * 16 * 64];  // 32 KB P@V partials (160 KB: 2 WG/WGP)

    const int blk = blockIdx.x;        // bh*128 + nt
    const int nt  = blk & 127;
    const int bh  = blk >> 7;          // b*H + h
    const int h   = bh % H_;
    const int b   = bh / H_;
    const int n0  = nt * 16;

    const int w    = threadIdx.x >> 5;
    const int lane = threadIdx.x & 31;
    const int lr   = lane & 15;
    const int g    = lane >> 4;

    // Q tile A-fragments (coalesced packed loads)
    v16h qa[2];
#pragma unroll
    for (int kc = 0; kc < 2; ++kc)
        qa[kc] = load_frag16(Qpk, ((size_t)bh * 128 + nt) * 2 + kc, lane);

    // --- Phase A: scores into LDS (each wave: 16 of 128 m-tiles) -----------
#pragma unroll 2
    for (int mtile = w; mtile < 128; mtile += 8) {
        v8f acc = (v8f){};
#pragma unroll
        for (int kc = 0; kc < 2; ++kc) {
            v16h kb = load_frag16(Kpk, ((size_t)bh * 128 + mtile) * 2 + kc, lane);
            acc = WMMA_F16(qa[kc], kb, acc);
        }
#pragma unroll
        for (int r = 0; r < 8; ++r) S[(r + 8 * g) * 2048 + mtile * 16 + lr] = acc[r];
    }
    __syncthreads();

    // --- Phase B: softmax over m; write probs to d_out attention region ----
    {
        const int row = threadIdx.x >> 4;  // 16 threads/row within a half-wave
        const int col = threadIdx.x & 15;
        float* srow = &S[row * 2048];

        float mx = -1e30f;
        for (int i = col; i < 2048; i += 16) mx = fmaxf(mx, srow[i]);
#pragma unroll
        for (int off = 8; off > 0; off >>= 1) mx = fmaxf(mx, __shfl_xor(mx, off, 16));

        float sum = 0.0f;
        for (int i = col; i < 2048; i += 16) {
            const float e = __expf(srow[i] - mx);
            srow[i] = e;
            sum += e;
        }
#pragma unroll
        for (int off = 8; off > 0; off >>= 1) sum += __shfl_xor(sum, off, 16);
        const float inv = 1.0f / sum;

        const long base = ((long)(b * N_ + n0 + row)) * N_ * H_ + h;
        for (int i = col; i < 2048; i += 16) {
            const float p = srow[i] * inv;
            srow[i] = p;
            attn_out[base + (long)i * H_] = p;  // attention[b, n, m, h]
        }
    }
    __syncthreads();

    // --- Phase C: O = P @ V (each wave: 8 of 64 k=32 m-chunks) -------------
    v8f oacc[4];
#pragma unroll
    for (int s = 0; s < 4; ++s) oacc[s] = (v8f){};

#pragma unroll 2
    for (int kc = w; kc < 64; kc += 8) {
        v16h pa;
        {
            const float* prow = &S[lr * 2048 + kc * 32 + 8 * g];
#pragma unroll
            for (int j = 0; j < 8; ++j) {
                pa[j]     = (_Float16)prow[j];
                pa[8 + j] = (_Float16)prow[16 + j];
            }
        }
#pragma unroll
        for (int s = 0; s < 4; ++s) {
            v16h vb = load_frag16(Vpk, ((size_t)bh * 4 + s) * 64 + kc, lane);
            oacc[s] = WMMA_F16(pa, vb, oacc[s]);
        }
    }
#pragma unroll
    for (int s = 0; s < 4; ++s)
#pragma unroll
        for (int r = 0; r < 8; ++r)
            Op[w * 1024 + (r + 8 * g) * 64 + s * 16 + lr] = oacc[s][r];
    __syncthreads();

    // reduce 8 wave partials; emit head output in proj kernel's packed A layout
    const int mtile3 = b * 128 + nt;   // (b*N + n0) / 16
    for (int i = threadIdx.x; i < 1024; i += 256) {
        float sum = 0.0f;
#pragma unroll
        for (int ww = 0; ww < 8; ++ww) sum += Op[ww * 1024 + i];
        const int nl = i >> 6, d = i & 63;
        const int c = h * D_ + d;
        const int ko = c & 31, gq = (ko >> 3) & 1;
        const int e = (ko & 7) + 8 * (ko >> 4);
        const size_t idx = (((size_t)mtile3 * 32 + (c >> 5)) * 32 +
                            (nl + 16 * gq)) * 16 + e;
        Opk[idx] = (_Float16)sum;
    }
}

// ---------------------------------------------------------------------------
// Kernel 3: output projection  out = O @ Wp.T + bp  (packed A and B, fp32 out)
// ---------------------------------------------------------------------------
__global__ __launch_bounds__(256) void proj_kernel(
    const _Float16* __restrict__ Opk, const _Float16* __restrict__ wppk,
    const float* __restrict__ bp, float* __restrict__ out)
{
    const int wid  = blockIdx.x * 8 + (threadIdx.x >> 5);
    const int lane = threadIdx.x & 31;
    const int lr   = lane & 15;
    const int g    = lane >> 4;
    const int cg   = wid % 16;   // 16 groups of 64 cols
    const int mt   = wid / 16;   // 512 row tiles

    v8f acc[4];
#pragma unroll
    for (int s = 0; s < 4; ++s) acc[s] = (v8f){};

    const size_t af0 = (size_t)mt * 32;
    size_t bf0[4];
#pragma unroll
    for (int s = 0; s < 4; ++s) bf0[s] = (size_t)(cg * 4 + s) * 32;

    gemm_k32(Opk, wppk, af0, bf0, lane, acc);

#pragma unroll
    for (int s = 0; s < 4; ++s) {
        const int c = cg * 64 + s * 16 + lr;
        const float bias = bp[c];
#pragma unroll
        for (int r = 0; r < 8; ++r)
            out[(size_t)(mt * 16 + r + 8 * g) * C_ + c] = acc[s][r] + bias;
    }
}

// ---------------------------------------------------------------------------
extern "C" void kernel_launch(void* const* d_in, const int* in_sizes, int n_in,
                              void* d_out, int out_size, void* d_ws, size_t ws_size,
                              hipStream_t stream)
{
    const float* x   = (const float*)d_in[0];
    const float* Wq  = (const float*)d_in[1];
    const float* bq  = (const float*)d_in[2];
    const float* Wkv = (const float*)d_in[3];
    const float* bkv = (const float*)d_in[4];
    const float* Wp  = (const float*)d_in[5];
    const float* bp  = (const float*)d_in[6];

    float* out      = (float*)d_out;                        // [B,N,C]
    float* attn_out = (float*)d_out + (size_t)B_ * N_ * C_; // [B,N,N,H]

    // Workspace layout (halves), ~92 MB total
    const size_t tok = (size_t)B_ * N_ * C_;  // 8,388,608
    _Float16* xpk  = (_Float16*)d_ws;         // packed A of x
    _Float16* wpk  = xpk + tok;               // packed B of [Wq;Wkv]: 3072*1024
    _Float16* wppk = wpk + (size_t)3072 * 1024;  // packed B of Wp: 1024*1024
    _Float16* Qpk  = wppk + (size_t)1024 * 1024;
    _Float16* Kpk  = Qpk + tok;
    _Float16* Vpk  = Kpk + tok;
    _Float16* Opk  = Vpk + tok;

    // Pack passes (one-time f32->f16 + fragment swizzle)
    pack_a_f32<<<2048, 256, 0, stream>>>(x, xpk, 32);              // 16384 frags
    pack_b_f32<<<256,  256, 0, stream>>>(Wq, wpk, 32);             // 2048 frags
    pack_b_f32<<<512,  256, 0, stream>>>(Wkv, wpk + (size_t)2048 * 512, 32);
    pack_b_f32<<<256,  256, 0, stream>>>(Wp, wppk, 32);            // 2048 frags

    // K1: 512 row tiles * 48 col groups = 24576 waves / 8 per block
    qkv_kernel<<<3072, 256, 0, stream>>>(xpk, wpk, bq, bkv, Qpk, Kpk, Vpk);
    // K2: one block per (b, h, 16-row tile)
    attn_kernel<<<8192, 256, 0, stream>>>(Qpk, Kpk, Vpk, attn_out, Opk);
    // K3: 512 row tiles * 16 col groups = 8192 waves / 8 per block
    proj_kernel<<<1024, 256, 0, stream>>>(Opk, wppk, bp, out);
}
